// RFFNet_33500744909020
// MI455X (gfx1250) — matrix operand order repaired and checked by reading, not compiled, not run on hardware
//
#include <hip/hip_runtime.h>
#include <stdint.h>

// Problem dims (fixed by reference): T=100, B=256, N=1024
#define T_STEPS 100
#define BATCH   256
#define NDIM    1024
#define MTOT    (T_STEPS * BATCH)   // 25600 GEMM rows

#define BLK_M   128                 // per-block M tile (8 waves x 16)
#define BLK_N   128                 // per-block N tile (8 wmma tiles per wave)
#define NT      (BLK_N / 16)        // 8 N-tiles per wave
#define LDSW_STRIDE 20              // dwords per LDS W row: 16 data + 4 pad (16B-aligned rows, conflict-free b128 reads)

typedef __attribute__((ext_vector_type(16))) __bf16   v16bf;
typedef __attribute__((ext_vector_type(2)))  __bf16   v2bf;
typedef __attribute__((ext_vector_type(8)))  float    v8f;
typedef __attribute__((ext_vector_type(4)))  float    v4f;
typedef __attribute__((ext_vector_type(2)))  float    v2f;
typedef __attribute__((ext_vector_type(16))) float    v16f;
typedef __attribute__((ext_vector_type(4)))  uint32_t v4u;

union AFrag { v16bf bf; uint32_t u[8]; };
union BFrag { v16bf bf; v4u q[2]; };

__device__ __forceinline__ void sched_fence() {
#if __has_builtin(__builtin_amdgcn_sched_barrier)
    __builtin_amdgcn_sched_barrier(0);   // pin the ds-load clause before the WMMA block
#else
    asm volatile("" ::: "memory");
#endif
}

// f32 pair -> packed bf16 dword (one v_cvt_pk_bf16_f32)
__device__ __forceinline__ uint32_t pack_bf16(float lo, float hi) {
    v2bf r = __builtin_convertvector((v2f){lo, hi}, v2bf);
    return __builtin_bit_cast(uint32_t, r);
}

// --- Kernel 1: W (f32, row-major [N][N]) -> packed bf16 pairs in workspace ---
__global__ void __launch_bounds__(256) convert_w(const float* __restrict__ W,
                                                 uint32_t* __restrict__ Wbf) {
    int i = blockIdx.x * 256 + threadIdx.x;      // one dword (2 elements) per thread
    v2f w = *(const v2f*)(W + 2 * i);
    Wbf[i] = pack_bf16(w[0], w[1]);
}

// --- Kernel 2: CUR = X @ W^T + b via v_wmma_f32_16x16x32_bf16 ---
// X: [MTOT, NDIM] f32 (x[T,B,N] flattened), Wbf: packed bf16 [NDIM][NDIM/2] dwords
// out: [MTOT, NDIM] f32 (d_out; overwritten by scan afterwards)
// grid: (NDIM/BLK_N, MTOT/BLK_M) — N fastest so consecutive blocks share one X slab (cache-hot)
__global__ void __launch_bounds__(256)
gemm_cur(const float* __restrict__ X, const uint32_t* __restrict__ Wbf,
         const float* __restrict__ bias, float* __restrict__ out) {
    __shared__ uint32_t ldsW[BLK_N * LDSW_STRIDE];   // 128 rows x 16 data dwords (+pad) = 10 KB

    const int tid   = threadIdx.x;
    const int lane  = tid & 31;
    const int wave  = tid >> 5;                       // 0..7
    const int nBase = blockIdx.x * BLK_N;
    const int rowG  = blockIdx.y * BLK_M + wave * 16 + (lane & 15); // A-matrix row (M)
    const int kbA   = (lane < 16) ? 0 : 8;            // A k sub-base (ISA A layout)
    const int kbBu  = (lane < 16) ? 0 : 4;            // B k sub-base in dwords (K=0 vs K=8)
    const int ldsB0 = (lane & 15) * LDSW_STRIDE + kbBu; // this lane's B-frag base (tile 0)

    v8f acc[NT];
    #pragma unroll
    for (int j = 0; j < NT; ++j)
        acc[j] = (v8f){0.f, 0.f, 0.f, 0.f, 0.f, 0.f, 0.f, 0.f};

    const float* xrow = X + (size_t)rowG * NDIM;

    // staging: 128 rows x 16 dwords = 2048 dwords; 256 threads x 2 v4u
    const int sr = tid >> 1;            // 0..127 (row within N tile)
    const int sc = (tid & 1) * 8;       // dword column: 0 or 8

    for (int kk = 0; kk < NDIM / 32; ++kk) {
        const int k0 = kk * 32;

        __syncthreads();
        // stage W tile (B-matrix rows = n, 32 contiguous bf16 of k) into LDS
        {
            const uint32_t* src = Wbf + (size_t)(nBase + sr) * (NDIM / 2) + (k0 / 2) + sc;
            uint32_t* dst = ldsW + sr * LDSW_STRIDE + sc;
            *(v4u*)(dst)     = *(const v4u*)(src);
            *(v4u*)(dst + 4) = *(const v4u*)(src + 4);
        }
        // prefetch next K-tile of X while LDS settles
        __builtin_prefetch(xrow + k0 + 32, 0, 0);
        __syncthreads();

        // A fragment: M = lane&15; K = {kbA..kbA+7} U {kbA+16..kbA+23}, even/odd packed.
        // Single <16 x f32> -> <16 x bf16> convert => 8x v_cvt_pk_bf16_f32, no half-moves.
        AFrag a;
        {
            v4f F0 = *(const v4f*)(xrow + k0 + kbA);
            v4f F1 = *(const v4f*)(xrow + k0 + kbA + 4);
            v4f F2 = *(const v4f*)(xrow + k0 + kbA + 16);
            v4f F3 = *(const v4f*)(xrow + k0 + kbA + 20);
            v16f f;
            f[0]=F0[0];  f[1]=F0[1];  f[2]=F0[2];  f[3]=F0[3];
            f[4]=F1[0];  f[5]=F1[1];  f[6]=F1[2];  f[7]=F1[3];
            f[8]=F2[0];  f[9]=F2[1];  f[10]=F2[2]; f[11]=F2[3];
            f[12]=F3[0]; f[13]=F3[1]; f[14]=F3[2]; f[15]=F3[3];
            a.bf = __builtin_convertvector(f, v16bf);
        }

        // load ALL 8 B fragments (one 16x ds_load_b128 clause)...
        BFrag bm[NT];
        #pragma unroll
        for (int j = 0; j < NT; ++j) {
            const uint32_t* p = ldsW + j * 16 * LDSW_STRIDE + ldsB0;
            bm[j].q[0] = *(const v4u*)(p);
            bm[j].q[1] = *(const v4u*)(p + 8);
        }
        // ...keep the scheduler from sinking them onto their consumers, so the
        // dscnt waits pipeline (<=14,12,...) instead of eight blocking waits of 0
        sched_fence();
        #pragma unroll
        for (int j = 0; j < NT; ++j) {
            acc[j] = __builtin_amdgcn_wmma_f32_16x16x32_bf16(
                /*neg_a=*/false, a.bf, /*neg_b=*/false, bm[j].bf,
                /*c_mod=*/(short)0, acc[j], /*reuse_a=*/false, /*reuse_b=*/false);
        }
    }

    // epilogue: C/D layout — N = lane&15, M = r + (lane<16 ? 0 : 8); fold bias
    const int mTop = blockIdx.y * BLK_M + wave * 16 + ((lane < 16) ? 0 : 8);
    #pragma unroll
    for (int j = 0; j < NT; ++j) {
        const int nCol = nBase + j * 16 + (lane & 15);
        const float bj = bias[nCol];
        #pragma unroll
        for (int r = 0; r < 8; ++r) {
            out[(size_t)(mTop + r) * NDIM + nCol] = acc[j][r] + bj;
        }
    }
}

// --- Kernel 3: elementwise leaky-integrate-fire scan over T, in place on d_out ---
__global__ void __launch_bounds__(256)
scan_spk(const float* __restrict__ X, const float* __restrict__ Vp,
         float* __restrict__ out) {
    const int idx = blockIdx.x * 256 + threadIdx.x;   // 0 .. B*N-1
    const float V = Vp[0];
    float spk = X[idx];                               // spk0 = x[0] (raw)
    float mem = 0.0f;
    const size_t BN = (size_t)BATCH * NDIM;
    size_t p = (size_t)idx;
    #pragma unroll 4
    for (int t = 0; t < T_STEPS; ++t, p += BN) {
        const float cur   = out[p];                   // read CUR before overwrite
        const float reset = (mem > 1.0f) ? 1.0f : 0.0f;
        mem = 0.9f * mem + cur + V * spk - reset;     // THRESH = 1.0
        spk = (mem > 1.0f) ? 1.0f : 0.0f;
        out[p] = spk;
    }
}

extern "C" void kernel_launch(void* const* d_in, const int* in_sizes, int n_in,
                              void* d_out, int out_size, void* d_ws, size_t ws_size,
                              hipStream_t stream) {
    const float* X    = (const float*)d_in[0];   // [T,B,N]
    const float* W    = (const float*)d_in[1];   // [N,N]
    const float* bias = (const float*)d_in[2];   // [N]
    const float* V    = (const float*)d_in[3];   // scalar
    float* out        = (float*)d_out;           // [T,B,N]
    uint32_t* Wbf     = (uint32_t*)d_ws;         // 2 MB: bf16 W

    (void)in_sizes; (void)n_in; (void)out_size; (void)ws_size;

    // W f32 -> bf16 pairs: 1024*1024/2 dwords
    convert_w<<<(NDIM * NDIM / 2) / 256, 256, 0, stream>>>(W, Wbf);

    // big GEMM: [25600,1024] x [1024,1024]^T -> d_out
    // N-dim fastest (blockIdx.x) so consecutive blocks reuse the same 0.5 MB X slab
    dim3 grid(NDIM / BLK_N, MTOT / BLK_M);       // 8 x 200
    gemm_cur<<<grid, 256, 0, stream>>>(X, Wbf, bias, out);

    // independent per-(b,n) recurrences, in place
    scan_spk<<<(BATCH * NDIM) / 256, 256, 0, stream>>>(X, V, out);
}
